// CausalSelfAttention_69526930587689
// MI455X (gfx1250) — matrix operand order repaired and checked by reading, not compiled, not run on hardware
//
#include <hip/hip_runtime.h>
#include <hip/hip_bf16.h>
#include <math.h>

// CausalSelfAttention for MI455X (gfx1250, wave32, WMMA).
// Pipeline: cvt(bf16) -> QKV GEMM (v_wmma_f32_16x16x32_bf16, fp32 acc,
//           B-tile double-buffered in LDS via global_load_async_to_lds_b128)
//           -> RoPE/pack -> flash attention (WMMA + online softmax)
//           -> output GEMM (fp32 out).
// Workspace layout (bytes), total ~288 MiB (see launcher).

typedef __attribute__((ext_vector_type(16))) __bf16 v16bf;
typedef __attribute__((ext_vector_type(4)))  __bf16 v4bf;
typedef __attribute__((ext_vector_type(8)))  float  v8f;
typedef int vec4i __attribute__((vector_size(16)));   // matches builtin param type

#define B_SZ   4
#define T_SZ   2048
#define C_SZ   2048
#define H_SZ   16
#define D_SZ   128
#define C3_SZ  6144
#define M_SZ   8192   // B*T

#define AS1 __attribute__((address_space(1)))
#define AS3 __attribute__((address_space(3)))

#if defined(__AMDGCN__) && __has_builtin(__builtin_amdgcn_global_load_async_to_lds_b128)
#define USE_ASYNC_LDS 1
#else
#define USE_ASYNC_LDS 0
#endif

#define LDSB_LD 136   // 128 + 8 pad (bf16 elems): de-conflicts lane-per-row reads

static __device__ __forceinline__ v8f wmma_bf16(v16bf a, v16bf b, v8f c) {
  return __builtin_amdgcn_wmma_f32_16x16x32_bf16(false, a, false, b, (short)0, c,
                                                 false, false);
}

// A-operand (16x32, MxK): lane&15 = M-row; regs 0-3 hold K = kb..kb+7,
// regs 4-7 hold K = kb+16..kb+23 with kb = (lane>>4)*8.  Two 16B loads.
static __device__ __forceinline__ v16bf load_frag_a(const __bf16* __restrict__ p,
                                                    int ld, int row0, int k0,
                                                    int lane) {
  const int m  = lane & 15;
  const int kb = (lane >> 4) << 3;
  const __bf16* base = p + (size_t)(row0 + m) * ld + (k0 + kb);
  union { v16bf v; uint4 u[2]; } r;
  r.u[0] = *(const uint4*)(base);
  r.u[1] = *(const uint4*)(base + 16);
  return r.v;
}

// B-operand (32x16, KxN) from row-major [K,N]: lane&15 = N-col; elems 0-7 hold
// K = kb..kb+7, elems 8-15 hold K = kb+16..kb+23 with kb = (lane>>4)*8.
static __device__ __forceinline__ v16bf load_frag_b(const __bf16* __restrict__ p,
                                                    int ldn, int k0, int n0,
                                                    int lane) {
  const __bf16* base = p + (size_t)(k0 + ((lane >> 4) << 3)) * ldn + n0 + (lane & 15);
  v16bf r;
#pragma unroll
  for (int i = 0; i < 8; ++i) {
    r[i]     = base[(size_t)i * ldn];
    r[i + 8] = base[(size_t)(i + 16) * ldn];
  }
  return r;
}

// Same B-operand pattern, but from an LDS tile [32][LDSB_LD].
static __device__ __forceinline__ v16bf load_frag_b_lds(const __bf16* lb,
                                                        int n0, int lane) {
  const __bf16* base = lb + ((lane >> 4) << 3) * LDSB_LD + n0 + (lane & 15);
  v16bf r;
#pragma unroll
  for (int i = 0; i < 8; ++i) {
    r[i]     = base[i * LDSB_LD];
    r[i + 8] = base[(i + 16) * LDSB_LD];
  }
  return r;
}

static __device__ __forceinline__ v8f zero8() {
  v8f z = {0.f, 0.f, 0.f, 0.f, 0.f, 0.f, 0.f, 0.f};
  return z;
}

// Stage one 32x128 B tile (rows k0..k0+31, cols nb0..nb0+127) into LDS.
// 256 threads: thread t copies row t>>3, 16 cols at (t&7)*16 (two b128).
static __device__ __forceinline__ void stage_b_tile(const __bf16* __restrict__ B,
                                                    int N, int k0, int nb0,
                                                    __bf16* lbuf, int tid) {
  const int crow = tid >> 3;
  const int ccol = (tid & 7) * 16;
  const __bf16* gp = B + (size_t)(k0 + crow) * N + nb0 + ccol;
  __bf16* lq = lbuf + crow * LDSB_LD + ccol;
#if USE_ASYNC_LDS
  __builtin_amdgcn_global_load_async_to_lds_b128((AS1 vec4i*)gp, (AS3 vec4i*)lq,
                                                 0, 0);
  __builtin_amdgcn_global_load_async_to_lds_b128((AS1 vec4i*)(gp + 8),
                                                 (AS3 vec4i*)(lq + 8), 0, 0);
#else
  *(uint4*)lq       = *(const uint4*)gp;
  *(uint4*)(lq + 8) = *(const uint4*)(gp + 8);
#endif
}

static __device__ __forceinline__ void wait_stage_barrier() {
#if USE_ASYNC_LDS
  asm volatile("s_wait_asynccnt 0x0" ::: "memory");
#endif
  __syncthreads();
}

// ---------------------------------------------------------------- fp32->bf16
__global__ void cvt_f32_bf16_kernel(const float* __restrict__ in,
                                    __bf16* __restrict__ out, int n4) {
  int i = blockIdx.x * blockDim.x + threadIdx.x;
  if (i < n4) {
    float4 f = ((const float4*)in)[i];
    v4bf o;
    o.x = (__bf16)f.x; o.y = (__bf16)f.y; o.z = (__bf16)f.z; o.w = (__bf16)f.w;
    ((v4bf*)out)[i] = o;
  }
}

// ----------------------------------------------------------- bf16 WMMA GEMM
// C[M,N] = A[M,K] @ B[K,N] + bias[N].  256 thr = 8 waves (4x2),
// block tile 128x128, wave tile 32x64, K-step 32.  B tile double-buffered in
// LDS (async copies); A fragments direct from global (L2-resident).
template <bool F32OUT>
__global__ __launch_bounds__(256)
void gemm_bf16_kernel(const __bf16* __restrict__ A, const __bf16* __restrict__ B,
                      const float* __restrict__ bias, void* __restrict__ Cout,
                      int M, int N, int K) {
  __shared__ __align__(16) __bf16 ldsB[2][32][LDSB_LD];

  const int tid  = threadIdx.x;
  const int lane = tid & 31;
  const int wid  = tid >> 5;
  const int wm   = wid >> 1;                    // 0..3
  const int wn   = wid & 1;                     // 0..1
  const int m0   = blockIdx.y * 128 + wm * 32;
  const int nb0  = blockIdx.x * 128;
  const int n0   = nb0 + wn * 64;

  v8f acc[2][4];
#pragma unroll
  for (int i = 0; i < 2; ++i)
#pragma unroll
    for (int j = 0; j < 4; ++j) acc[i][j] = zero8();

  stage_b_tile(B, N, 0, nb0, &ldsB[0][0][0], tid);
  wait_stage_barrier();

  int it = 0;
  for (int k0 = 0; k0 < K; k0 += 32, ++it) {
    const int cur = it & 1;
    if (k0 + 32 < K)
      stage_b_tile(B, N, k0 + 32, nb0, &ldsB[cur ^ 1][0][0], tid);

    v16bf a0 = load_frag_a(A, K, m0,      k0, lane);
    v16bf a1 = load_frag_a(A, K, m0 + 16, k0, lane);
    const __bf16* lb = &ldsB[cur][0][0];
    v16bf b[4];
#pragma unroll
    for (int j = 0; j < 4; ++j) b[j] = load_frag_b_lds(lb, wn * 64 + j * 16, lane);
#pragma unroll
    for (int j = 0; j < 4; ++j) {
      acc[0][j] = wmma_bf16(a0, b[j], acc[0][j]);
      acc[1][j] = wmma_bf16(a1, b[j], acc[1][j]);
    }
    wait_stage_barrier();   // next tile landed; everyone done reading cur
  }

  const int ncol  = lane & 15;
  const int nhalf = (lane >> 4) << 3;
#pragma unroll
  for (int i = 0; i < 2; ++i) {
    const size_t rowbase = (size_t)(m0 + i * 16 + nhalf) * N;
#pragma unroll
    for (int j = 0; j < 4; ++j) {
      const int n = n0 + j * 16 + ncol;
      const float bv = bias[n];
#pragma unroll
      for (int r = 0; r < 8; ++r) {
        const float val = acc[i][j][r] + bv;
        const size_t idx = rowbase + (size_t)r * N + n;
        if (F32OUT) ((float*)Cout)[idx] = val;
        else        ((__bf16*)Cout)[idx] = (__bf16)val;
      }
    }
  }
}

// ------------------------------------------------------------ RoPE + repack
// thread = (b,h,t,j), j in [0,64): handles dims (j, j+64) of one head.
__global__ void rope_pack_kernel(const __bf16* __restrict__ qkv,
                                 __bf16* __restrict__ q,
                                 __bf16* __restrict__ kT,
                                 __bf16* __restrict__ v) {
  const int idx = blockIdx.x * blockDim.x + threadIdx.x;
  const int j = idx & 63;
  const int t = (idx >> 6) & (T_SZ - 1);
  const int h = (idx >> 17) & (H_SZ - 1);
  const int b = idx >> 21;

  const size_t row = (size_t)(b * T_SZ + t) * C3_SZ;
  const int cb = h * D_SZ;
  const float q1 = (float)qkv[row + cb + j];
  const float q2 = (float)qkv[row + cb + j + 64];
  const float k1 = (float)qkv[row + C_SZ + cb + j];
  const float k2 = (float)qkv[row + C_SZ + cb + j + 64];
  const float v1 = (float)qkv[row + 2 * C_SZ + cb + j];
  const float v2 = (float)qkv[row + 2 * C_SZ + cb + j + 64];

  // inv_freq = 10000^(-j/64) ; ln(10000)/64 = 0.14391156831212787
  const float ang = (float)t * __expf(-0.14391157f * (float)j);
  float s, c;
  __sincosf(ang, &s, &c);

  const float scale = 0.08838834764831845f;  // 1/sqrt(128)
  const float qo1 = (q1 * c - q2 * s) * scale;
  const float qo2 = (q2 * c + q1 * s) * scale;
  const float ko1 = k1 * c - k2 * s;
  const float ko2 = k2 * c + k1 * s;

  const size_t bh = (size_t)(b * H_SZ + h);
  const size_t qi = (bh * T_SZ + t) * D_SZ;
  q[qi + j]      = (__bf16)qo1;
  q[qi + j + 64] = (__bf16)qo2;
  v[qi + j]      = (__bf16)v1;
  v[qi + j + 64] = (__bf16)v2;
  const size_t kbase = bh * (size_t)(D_SZ * T_SZ);
  kT[kbase + (size_t)j * T_SZ + t]        = (__bf16)ko1;
  kT[kbase + (size_t)(j + 64) * T_SZ + t] = (__bf16)ko2;
}

// ------------------------------------------------------- flash attention
// grid = (T/64, B*H), 128 thr = 4 waves. Wave owns 16 query rows, streams
// 64-wide key tiles; online softmax in fp32; P via per-wave LDS tile
// (C-layout -> A-layout); PV accumulated in 8 f32 WMMA tiles.
__global__ __launch_bounds__(128)
void flash_attn_kernel(const __bf16* __restrict__ q,
                       const __bf16* __restrict__ kT,
                       const __bf16* __restrict__ v,
                       __bf16* __restrict__ y) {
  __shared__ __align__(16) __bf16 ldsP[4][16][72];   // +8 pad: conflict-free reads

  const int lane = threadIdx.x & 31;
  const int wid  = threadIdx.x >> 5;
  const int bh   = blockIdx.y;
  const int b    = bh >> 4;
  const int h    = bh & 15;
  const int t0w  = blockIdx.x * 64 + wid * 16;       // first query row of wave

  const __bf16* qbh = q  + (size_t)bh * T_SZ * D_SZ;
  const __bf16* kbh = kT + (size_t)bh * D_SZ * T_SZ;
  const __bf16* vbh = v  + (size_t)bh * T_SZ * D_SZ;

  v16bf qf[4];
#pragma unroll
  for (int kk = 0; kk < 4; ++kk) qf[kk] = load_frag_a(qbh, D_SZ, t0w, kk * 32, lane);

  v8f oacc[8];
#pragma unroll
  for (int nt = 0; nt < 8; ++nt) oacc[nt] = zero8();
  float mrow[8], lrow[8];
#pragma unroll
  for (int r = 0; r < 8; ++r) { mrow[r] = -__builtin_inff(); lrow[r] = 0.f; }

  const int ncol  = lane & 15;
  const int nhalf = (lane >> 4) << 3;

  for (int s0 = 0; s0 <= t0w + 15; s0 += 64) {
    // ---- S = Q @ K^T (16 x 64), fp32 accum
    v8f sacc[4];
#pragma unroll
    for (int j = 0; j < 4; ++j) {
      v8f sa = zero8();
#pragma unroll
      for (int kk = 0; kk < 4; ++kk) {
        v16bf bf = load_frag_b(kbh, T_SZ, kk * 32, s0 + j * 16, lane);
        sa = wmma_bf16(qf[kk], bf, sa);
      }
      sacc[j] = sa;
    }

    const bool domask = (s0 + 63 > t0w);
    float alpha[8];
#pragma unroll
    for (int r = 0; r < 8; ++r) {
      const int trow = t0w + r + nhalf;
      if (domask) {
#pragma unroll
        for (int j = 0; j < 4; ++j) {
          const int scol = s0 + j * 16 + ncol;
          if (scol > trow) sacc[j][r] = -__builtin_inff();
        }
      }
      // row max over 64 cols: 4 local + xor-reduce over 16-lane half
      float vmax = fmaxf(fmaxf(sacc[0][r], sacc[1][r]),
                         fmaxf(sacc[2][r], sacc[3][r]));
      for (int off = 1; off < 16; off <<= 1)
        vmax = fmaxf(vmax, __shfl_xor(vmax, off, 32));
      const float mnew = fmaxf(mrow[r], vmax);
      const float al   = __expf(mrow[r] - mnew);   // first tile: exp(-inf)=0
      mrow[r]  = mnew;
      alpha[r] = al;

      float ps = 0.f;
#pragma unroll
      for (int j = 0; j < 4; ++j) {
        const float p = __expf(sacc[j][r] - mnew);
        ldsP[wid][r + nhalf][j * 16 + ncol] = (__bf16)p;
        ps += p;
      }
      for (int off = 1; off < 16; off <<= 1) ps += __shfl_xor(ps, off, 32);
      lrow[r] = lrow[r] * al + ps;
    }

    // rescale running output
#pragma unroll
    for (int nt = 0; nt < 8; ++nt)
#pragma unroll
      for (int r = 0; r < 8; ++r) oacc[nt][r] *= alpha[r];

    asm volatile("s_wait_dscnt 0" ::: "memory");   // P stores visible (wave-local)

    // ---- O += P @ V  (P as A-operand from LDS; V row-major is B-operand)
    v16bf pf0 = load_frag_a(&ldsP[wid][0][0], 72, 0, 0,  lane);
    v16bf pf1 = load_frag_a(&ldsP[wid][0][0], 72, 0, 32, lane);
#pragma unroll
    for (int nt = 0; nt < 8; ++nt) {
      v16bf b0 = load_frag_b(vbh, D_SZ, s0,      nt * 16, lane);
      v16bf b1 = load_frag_b(vbh, D_SZ, s0 + 32, nt * 16, lane);
      oacc[nt] = wmma_bf16(pf0, b0, oacc[nt]);
      oacc[nt] = wmma_bf16(pf1, b1, oacc[nt]);
    }
  }

  // normalize + write y[B,T,C] (c = h*128 + d) as bf16 for the output GEMM
#pragma unroll
  for (int r = 0; r < 8; ++r) {
    const float inv = 1.0f / lrow[r];
    const int t = t0w + r + nhalf;
    const size_t yrow = ((size_t)(b * T_SZ + t)) * C_SZ + h * D_SZ;
#pragma unroll
    for (int nt = 0; nt < 8; ++nt)
      y[yrow + nt * 16 + ncol] = (__bf16)(oacc[nt][r] * inv);
  }
}

// --------------------------------------------------------------- launcher
extern "C" void kernel_launch(void* const* d_in, const int* in_sizes, int n_in,
                              void* d_out, int out_size, void* d_ws, size_t ws_size,
                              hipStream_t stream) {
  const float* x      = (const float*)d_in[0];
  const float* W_attn = (const float*)d_in[1];
  const float* b_attn = (const float*)d_in[2];
  const float* W_proj = (const float*)d_in[3];
  const float* b_proj = (const float*)d_in[4];
  float* out = (float*)d_out;

  char* ws = (char*)d_ws;
  __bf16* xbf   = (__bf16*)(ws);                       // 32 MiB
  __bf16* WaBf  = (__bf16*)(ws + (size_t)33554432);    // 24 MiB
  __bf16* WpBf  = (__bf16*)(ws + (size_t)58720256);    //  8 MiB
  __bf16* qkvBf = (__bf16*)(ws + (size_t)67108864);    // 96 MiB
  __bf16* qBf   = (__bf16*)(ws + (size_t)167772160);   // 32 MiB
  __bf16* kTBf  = (__bf16*)(ws + (size_t)201326592);   // 32 MiB
  __bf16* vBf   = (__bf16*)(ws + (size_t)234881024);   // 32 MiB
  __bf16* yBf   = (__bf16*)(ws + (size_t)268435456);   // 32 MiB
  // requires ws_size >= 301989888 bytes (~288 MiB)

  // 1) converts
  cvt_f32_bf16_kernel<<<(M_SZ * C_SZ / 4 + 255) / 256, 256, 0, stream>>>(x, xbf, M_SZ * C_SZ / 4);
  cvt_f32_bf16_kernel<<<(C_SZ * C3_SZ / 4 + 255) / 256, 256, 0, stream>>>(W_attn, WaBf, C_SZ * C3_SZ / 4);
  cvt_f32_bf16_kernel<<<(C_SZ * C_SZ / 4 + 255) / 256, 256, 0, stream>>>(W_proj, WpBf, C_SZ * C_SZ / 4);

  // 2) qkv = x @ W_attn + b_attn  (bf16 out)
  gemm_bf16_kernel<false><<<dim3(C3_SZ / 128, M_SZ / 128), 256, 0, stream>>>(
      xbf, WaBf, b_attn, qkvBf, M_SZ, C3_SZ, C_SZ);

  // 3) RoPE + repack (q scaled, kT transposed)
  rope_pack_kernel<<<(B_SZ * H_SZ * T_SZ * 64) / 256, 256, 0, stream>>>(
      qkvBf, qBf, kTBf, vBf);

  // 4) causal flash attention
  flash_attn_kernel<<<dim3(T_SZ / 64, B_SZ * H_SZ), 128, 0, stream>>>(
      qBf, kTBf, vBf, yBf);

  // 5) out = y @ W_proj + b_proj  (fp32 out)
  gemm_bf16_kernel<true><<<dim3(C_SZ / 128, M_SZ / 128), 256, 0, stream>>>(
      yBf, WpBf, b_proj, out, M_SZ, C_SZ, C_SZ);
}
